// AttentionLayer_48086453846445
// MI455X (gfx1250) — compile-verified, matrix operand order checked by reading, not run on hardware
//
#include <hip/hip_runtime.h>
#include <hip/hip_bf16.h>

typedef __attribute__((ext_vector_type(16))) _Float16 v16h;
typedef __attribute__((ext_vector_type(8)))  _Float16 v8h;
typedef __attribute__((ext_vector_type(8)))  float    v8f;
typedef __attribute__((ext_vector_type(4)))  unsigned int u32x4;
typedef __attribute__((ext_vector_type(8)))  int      i32x8;
typedef __attribute__((ext_vector_type(4)))  int      i32x4;

#define D_MODEL 1024
#define N_HEADS 16
#define HEAD_DIM 64
#define SEQ_T 2048
#define BATCH 4
#define M_ROWS (BATCH * SEQ_T)      // 8192
#define N_QKV (3 * D_MODEL)         // 3072

#if __has_builtin(__builtin_amdgcn_tensor_load_to_lds) && \
    __has_builtin(__builtin_amdgcn_s_wait_tensorcnt)
#define HAVE_TDM 1
#else
#define HAVE_TDM 0
#endif

static __device__ __forceinline__ v8f wmma_f16(v16h a, v16h b, v8f c) {
    // D = A(16x32 f16) * B(32x16 f16) + C(16x16 f32)
    return __builtin_amdgcn_wmma_f32_16x16x32_f16(false, a, false, b, (short)0, c,
                                                  false, false);
}

// Build a 16-half A/B fragment from two contiguous 8-half groups at p and p+16.
static __device__ __forceinline__ v16h frag16(const _Float16* p) {
    v8h lo = *(const v8h*)p;
    v8h hi = *(const v8h*)(p + 16);
    v16h r;
#pragma unroll
    for (int i = 0; i < 8; ++i) { r[i] = lo[i]; r[i + 8] = hi[i]; }
    return r;
}

#if HAVE_TDM
// Issue a TDM load of a 2D tile (tile_k cols x tile_m rows) of 16-bit elements
// from a row-major [rows, row_len] tensor into LDS at lds_off (packed row-major).
// D# layout per CDNA5 ISA.
static __device__ __forceinline__ void tdm_load_tile_2d(
        unsigned lds_off, const void* gtile, unsigned row_len_elems,
        unsigned rows, unsigned tile_k, unsigned tile_m) {
    unsigned long long ga = (unsigned long long)gtile;
    u32x4 g0;
    g0[0] = 1u;                                   // count=1, no gather
    g0[1] = lds_off;                              // LDS byte address
    g0[2] = (unsigned)(ga & 0xFFFFFFFFu);         // global addr [31:0]
    g0[3] = (unsigned)((ga >> 32) & 0x01FFFFFFu)  // global addr [56:32]
            | (2u << 30);                         // type = 2 ("image")
    i32x8 g1;
    g1[0] = (int)(1u << 16);                      // data_size = 1 -> 2 bytes
    g1[1] = (int)((row_len_elems & 0xFFFFu) << 16);       // tensor_dim0 lo16
    g1[2] = (int)((row_len_elems >> 16) | ((rows & 0xFFFFu) << 16)); // dim0 hi|dim1 lo
    g1[3] = (int)((rows >> 16) | (tile_k << 16)); // dim1 hi | tile_dim0
    g1[4] = (int)tile_m;                          // tile_dim1 | tile_dim2=0
    g1[5] = (int)row_len_elems;                   // tensor_dim0_stride [31:0]
    g1[6] = 0;                                    // stride hi | dim1_stride lo
    g1[7] = 0;
    i32x4 z = {0, 0, 0, 0};
#if defined(__clang_major__) && __clang_major__ >= 23
    i32x8 z8 = {0, 0, 0, 0, 0, 0, 0, 0};
    __builtin_amdgcn_tensor_load_to_lds(g0, g1, z, z, z8, 0);
#else
    __builtin_amdgcn_tensor_load_to_lds(g0, g1, z, z, 0);
#endif
}
#endif

// ---------------------------------------------------------------- cast f32->f16
__global__ void cast_f32_to_f16_v8(const float* __restrict__ src,
                                   _Float16* __restrict__ dst, int n8) {
    int i = blockIdx.x * blockDim.x + threadIdx.x;
    if (i >= n8) return;
    size_t base = (size_t)i * 8;
    float4 a = *(const float4*)(src + base);
    float4 b = *(const float4*)(src + base + 4);
    v8h o;
    o[0] = (_Float16)a.x; o[1] = (_Float16)a.y;
    o[2] = (_Float16)a.z; o[3] = (_Float16)a.w;
    o[4] = (_Float16)b.x; o[5] = (_Float16)b.y;
    o[6] = (_Float16)b.z; o[7] = (_Float16)b.w;
    *(v8h*)(dst + base) = o;
}

// ------------------------------------- cast+transpose f32 [R,C] -> f16 [C,R]
// LDS-tiled 64x64 so both global sides are coalesced. grid (R/64, C/64).
__global__ __launch_bounds__(256)
void cast_transpose_f16(const float* __restrict__ src, _Float16* __restrict__ dst,
                        int R, int C) {
    __shared__ _Float16 tile[64][72];   // +8 halves pad: conflict-free gathers
    const int r0 = blockIdx.x * 64, c0 = blockIdx.y * 64;
    const int tid = threadIdx.x;
#pragma unroll
    for (int r = 0; r < 2; ++r) {
        int u = tid * 2 + r;            // 512 x v8h
        int rr = u >> 3, cc = (u & 7) * 8;
        const float* sp = src + (size_t)(r0 + rr) * C + c0 + cc;
        float4 a = *(const float4*)sp;
        float4 b = *(const float4*)(sp + 4);
        v8h o;
        o[0] = (_Float16)a.x; o[1] = (_Float16)a.y;
        o[2] = (_Float16)a.z; o[3] = (_Float16)a.w;
        o[4] = (_Float16)b.x; o[5] = (_Float16)b.y;
        o[6] = (_Float16)b.z; o[7] = (_Float16)b.w;
        *(v8h*)&tile[rr][cc] = o;
    }
    __syncthreads();
#pragma unroll
    for (int r = 0; r < 2; ++r) {
        int u = tid * 2 + r;
        int cr = u >> 3, rc = (u & 7) * 8;
        v8h o;
#pragma unroll
        for (int e = 0; e < 8; ++e) o[e] = tile[rc + e][cr];
        *(v8h*)(dst + (size_t)(c0 + cr) * R + r0 + rc) = o;
    }
}

// ------------------------------------------------- tiled WMMA GEMM  C = A*Wt^T + b
// A: [M,Kd] f16 row-major.  Wt: [N,Kd] f16 row-major (pre-transposed weights).
// Both LDS tiles are dense row-major 2D tiles -> both loaded by the TDM,
// double-buffered so the DMA overlaps the WMMA stream.
#define BM 128
#define BN 128
#define BK 32
template <typename OutT>
__global__ __launch_bounds__(256)
void gemm_wmma_f16(const _Float16* __restrict__ A, const _Float16* __restrict__ Wt,
                   const float* __restrict__ bias, OutT* __restrict__ Cout,
                   int Mdim, int Ndim, int Kdim) {
    __shared__ _Float16 lA[2][BM][BK];
    __shared__ _Float16 lB[2][BN][BK];

    const int tid  = threadIdx.x;
    const int wave = tid >> 5;
    const int lane = tid & 31;
    const int l16  = lane & 15;
    const int off  = (lane & 16) ? 8 : 0;
    const int half8 = (lane & 16) ? 8 : 0;
    const int wm   = wave >> 1;      // 0..3 -> 32 rows each
    const int wn   = wave & 1;       // 0..1 -> 64 cols each
    const int row0 = blockIdx.x * BM;
    const int n0   = blockIdx.y * BN;

    v8f acc[2][4];
#pragma unroll
    for (int mi = 0; mi < 2; ++mi)
#pragma unroll
        for (int ni = 0; ni < 4; ++ni) acc[mi][ni] = {};

#if HAVE_TDM
    unsigned ldsA[2], ldsB[2];
#pragma unroll
    for (int b = 0; b < 2; ++b) {
        ldsA[b] = (unsigned)(size_t)(void*)&lA[b][0][0];
        ldsB[b] = (unsigned)(size_t)(void*)&lB[b][0][0];
    }
#endif

    auto stage = [&](int buf, int kk) {
#if HAVE_TDM
        if (wave == 0) {
            tdm_load_tile_2d(ldsA[buf], A + (size_t)row0 * Kdim + kk,
                             (unsigned)Kdim, (unsigned)Mdim, BK, BM);
            tdm_load_tile_2d(ldsB[buf], Wt + (size_t)n0 * Kdim + kk,
                             (unsigned)Kdim, (unsigned)Ndim, BK, BN);
        }
#else
#pragma unroll
        for (int r = 0; r < 2; ++r) {
            int u = tid * 2 + r;      // 512 x v8h per tile
            int ar = u >> 2;
            int ac = (u & 3) * 8;
            *(v8h*)(&lA[buf][ar][ac]) =
                *(const v8h*)(A + (size_t)(row0 + ar) * Kdim + kk + ac);
            *(v8h*)(&lB[buf][ar][ac]) =
                *(const v8h*)(Wt + (size_t)(n0 + ar) * Kdim + kk + ac);
        }
#endif
    };
    auto stage_sync = [&]() {
#if HAVE_TDM
        if (wave == 0) __builtin_amdgcn_s_wait_tensorcnt(0);
#endif
        __syncthreads();
    };

    const int nsteps = Kdim / BK;
    int cur = 0;
    stage(0, 0);
    stage_sync();
    for (int s = 0; s < nsteps; ++s) {
        const int kk = s * BK;
        const bool more = (s + 1) < nsteps;
        if (more) stage(cur ^ 1, kk + BK);    // async DMA of next tile pair

        v16h af[2], bf[4];
#pragma unroll
        for (int mi = 0; mi < 2; ++mi)
            af[mi] = frag16(&lA[cur][wm * 32 + mi * 16 + l16][off]);
#pragma unroll
        for (int ni = 0; ni < 4; ++ni)
            bf[ni] = frag16(&lB[cur][wn * 64 + ni * 16 + l16][off]);
#pragma unroll
        for (int mi = 0; mi < 2; ++mi)
#pragma unroll
            for (int ni = 0; ni < 4; ++ni)
                acc[mi][ni] = wmma_f16(af[mi], bf[ni], acc[mi][ni]);

        if (more) stage_sync();
        cur ^= 1;
    }

    // ---- epilogue: branchless, hoisted base address, 64B contiguous per half
    const int colbase = n0 + wn * 64 + l16;
    OutT* outp = Cout + (size_t)(row0 + wm * 32 + half8) * Ndim + colbase;
#pragma unroll
    for (int mi = 0; mi < 2; ++mi)
#pragma unroll
        for (int ni = 0; ni < 4; ++ni) {
            const float bco = bias[colbase + ni * 16];
#pragma unroll
            for (int j = 0; j < 8; ++j)
                outp[(size_t)(mi * 16 + j) * Ndim + ni * 16] =
                    (OutT)(acc[mi][ni][j] + bco);
        }
}

// --------------------------------------- split qkv q/k parts -> [B,H,T,Dh]
__global__ void split_qk(const _Float16* __restrict__ qkv,
                         _Float16* __restrict__ q, _Float16* __restrict__ k) {
    int i = blockIdx.x * blockDim.x + threadIdx.x;
    if (i >= M_ROWS * 2 * D_MODEL / 8) return;
    int m  = i >> 8;                  // 256 v8h in the q|k span of each row
    int rr = (i & 255) * 8;
    int which = rr >> 10;             // 0=q 1=k
    int c = rr & 1023;
    int h = c >> 6;
    int d = c & 63;
    int b = m >> 11;
    int t = m & 2047;
    size_t dst = ((size_t)(b * N_HEADS + h) * SEQ_T + t) * HEAD_DIM + d;
    v8h val = *(const v8h*)(qkv + (size_t)m * N_QKV + rr);
    if (which == 0) *(v8h*)(q + dst) = val;
    else            *(v8h*)(k + dst) = val;
}

// ------------------------- V part of qkv -> transposed vt [B,H,Dh,T] (LDS tiled)
// grid (B*H, T/64), block 256
__global__ __launch_bounds__(256)
void transpose_v(const _Float16* __restrict__ qkv, _Float16* __restrict__ vt) {
    __shared__ _Float16 tile[64][72];
    const int bh = blockIdx.x;
    const int t0 = blockIdx.y * 64;
    const int b = bh >> 4, h = bh & 15;
    const int tid = threadIdx.x;
#pragma unroll
    for (int r = 0; r < 2; ++r) {
        int u = tid * 2 + r;
        int tt = u >> 3, dc = (u & 7) * 8;
        *(v8h*)&tile[tt][dc] = *(const v8h*)(
            qkv + (size_t)(b * SEQ_T + t0 + tt) * N_QKV + 2 * D_MODEL + h * 64 + dc);
    }
    __syncthreads();
#pragma unroll
    for (int r = 0; r < 2; ++r) {
        int u = tid * 2 + r;
        int dr = u >> 3, tc = (u & 7) * 8;
        v8h o;
#pragma unroll
        for (int e = 0; e < 8; ++e) o[e] = tile[tc + e][dr];
        *(v8h*)(vt + ((size_t)bh * HEAD_DIM + dr) * SEQ_T + t0 + tc) = o;
    }
}

// ------------------------------------------------- flash attention (causal)
// grid: (B*H, T/128); block 128 threads = 4 waves; 1 wave = 32 query rows
// (two 16-row WMMA tiles sharing every K and V fragment). Only the single
// diagonal key block is masked. Row-sums run on the matrix pipe (P x ones);
// the running-max butterfly is batched level-by-level so all 16 bpermutes of
// a level are in flight behind a single DScnt wait.
__global__ __launch_bounds__(128)
void attn_wmma(const _Float16* __restrict__ qh, const _Float16* __restrict__ kh,
               const _Float16* __restrict__ vt, _Float16* __restrict__ out) {
    __shared__ _Float16 sp[4][32 * 32];   // per-wave P C->A transpose tile

    const int bh   = blockIdx.x;
    const int wave = threadIdx.x >> 5;
    const int lane = threadIdx.x & 31;
    const int l16  = lane & 15;
    const int off  = (lane & 16) ? 8 : 0;
    const int half8 = (lane & 16) ? 8 : 0;
    const int qw   = blockIdx.y * 128 + wave * 32;

    const _Float16* Qp = qh + (size_t)bh * SEQ_T * HEAD_DIM;
    const _Float16* Kp = kh + (size_t)bh * SEQ_T * HEAD_DIM;
    const _Float16* Vt = vt + (size_t)bh * HEAD_DIM * SEQ_T;
    _Float16* lP = &sp[wave][0];

    // all-ones B fragment: rowsum(P) = P x 1 on the matrix pipe
    v16h ones;
#pragma unroll
    for (int i = 0; i < 16; ++i) ones[i] = (_Float16)1.0f;

    // Q fragments [mtile][k-chunk], pre-scaled by 1/sqrt(64) = 0.125
    v16h qf[2][2];
#pragma unroll
    for (int mt = 0; mt < 2; ++mt)
#pragma unroll
        for (int c = 0; c < 2; ++c) {
            v16h t = frag16(Qp + (size_t)(qw + mt * 16 + l16) * HEAD_DIM +
                            c * 32 + off);
#pragma unroll
            for (int i = 0; i < 16; ++i) t[i] = t[i] * (_Float16)0.125f;
            qf[mt][c] = t;
        }

    v8f O[2][4]; float mrow[2][8], lrow[2][8];
#pragma unroll
    for (int mt = 0; mt < 2; ++mt) {
#pragma unroll
        for (int t = 0; t < 4; ++t) O[mt][t] = {};
#pragma unroll
        for (int j = 0; j < 8; ++j) { mrow[mt][j] = -3.0e38f; lrow[mt][j] = 0.0f; }
    }

    auto kblock = [&](int keybase, bool masked) {
        // K fragments [key-half][k-chunk], shared by both query tiles
        v16h kf[2][2];
#pragma unroll
        for (int hb = 0; hb < 2; ++hb)
#pragma unroll
            for (int c = 0; c < 2; ++c)
                kf[hb][c] = frag16(Kp + (size_t)(keybase + hb * 16 + l16) *
                                   HEAD_DIM + c * 32 + off);
        // S[mtile][key-half]
        v8f S[2][2];
#pragma unroll
        for (int mt = 0; mt < 2; ++mt)
#pragma unroll
            for (int hb = 0; hb < 2; ++hb) {
                v8f s = {};
                s = wmma_f16(qf[mt][0], kf[hb][0], s);
                s = wmma_f16(qf[mt][1], kf[hb][1], s);
                S[mt][hb] = s;
            }
        // causal mask in place (diagonal block only)
        if (masked) {
#pragma unroll
            for (int mt = 0; mt < 2; ++mt)
#pragma unroll
                for (int j = 0; j < 8; ++j) {
                    const int rq = qw + mt * 16 + j + half8;
                    if (keybase + l16      > rq) S[mt][0][j] = -3.0e38f;
                    if (keybase + 16 + l16 > rq) S[mt][1][j] = -3.0e38f;
                }
        }
        // per-row local max, then batched XOR butterfly over 16-lane halves:
        // all 16 bpermutes of a level issue together behind one DScnt wait
        float mx[2][8];
#pragma unroll
        for (int mt = 0; mt < 2; ++mt)
#pragma unroll
            for (int j = 0; j < 8; ++j)
                mx[mt][j] = fmaxf(S[mt][0][j], S[mt][1][j]);
#pragma unroll
        for (int d = 1; d < 16; d <<= 1) {
            float sh[2][8];
#pragma unroll
            for (int mt = 0; mt < 2; ++mt)
#pragma unroll
                for (int j = 0; j < 8; ++j)
                    sh[mt][j] = __shfl_xor(mx[mt][j], d, 32);
#pragma unroll
            for (int mt = 0; mt < 2; ++mt)
#pragma unroll
                for (int j = 0; j < 8; ++j)
                    mx[mt][j] = fmaxf(mx[mt][j], sh[mt][j]);
        }
        // online softmax update + P write
        float alv[2][8];
#pragma unroll
        for (int mt = 0; mt < 2; ++mt)
#pragma unroll
            for (int j = 0; j < 8; ++j) {
                const float mn = fmaxf(mrow[mt][j], mx[mt][j]);
                const float al = __expf(mrow[mt][j] - mn);
                alv[mt][j] = al;
                mrow[mt][j] = mn;
                const float p0 = __expf(S[mt][0][j] - mn);
                const float p1 = __expf(S[mt][1][j] - mn);
#pragma unroll
                for (int t = 0; t < 4; ++t) O[mt][t][j] *= al;
                // P to LDS row-major [query 0..31][key 0..31]
                lP[(mt * 16 + j + half8) * 32 + l16]      = (_Float16)p0;
                lP[(mt * 16 + j + half8) * 32 + 16 + l16] = (_Float16)p1;
            }
        // reload P as A fragments (same-wave LDS ops are in-order: no barrier)
        v16h pf[2];
        pf[0] = frag16(&lP[l16 * 32 + off]);
        pf[1] = frag16(&lP[(16 + l16) * 32 + off]);
        // row sums on the matrix pipe: every output column holds rowsum(P)
        v8f rs0 = {}, rs1 = {};
        rs0 = wmma_f16(pf[0], ones, rs0);
        rs1 = wmma_f16(pf[1], ones, rs1);
#pragma unroll
        for (int j = 0; j < 8; ++j) {
            lrow[0][j] = lrow[0][j] * alv[0][j] + rs0[j];
            lrow[1][j] = lrow[1][j] * alv[1][j] + rs1[j];
        }
        // O += P * V : V^T rows contiguous in key; V fragment shared by tiles
#pragma unroll
        for (int t = 0; t < 4; ++t) {
            v16h vf = frag16(Vt + (size_t)(t * 16 + l16) * SEQ_T + keybase + off);
            O[0][t] = wmma_f16(pf[0], vf, O[0][t]);
            O[1][t] = wmma_f16(pf[1], vf, O[1][t]);
        }
    };

    const int kbdiag = qw >> 5;            // only this block straddles the diagonal
    for (int kb = 0; kb < kbdiag; ++kb) kblock(kb * 32, false);
    kblock(kbdiag * 32, true);

    // normalize + store into [B,T,C] f16
    const int b = bh >> 4;
    const int h = bh & 15;
#pragma unroll
    for (int mt = 0; mt < 2; ++mt)
#pragma unroll
        for (int j = 0; j < 8; ++j) {
            const float rl = 1.0f / lrow[mt][j];
            const int row = qw + mt * 16 + j + half8;
#pragma unroll
            for (int t = 0; t < 4; ++t) {
                int col = h * 64 + t * 16 + l16;
                out[((size_t)(b * SEQ_T + row)) * D_MODEL + col] =
                    (_Float16)(O[mt][t][j] * rl);
            }
        }
}

// ---------------------------------------------------------------- launch
extern "C" void kernel_launch(void* const* d_in, const int* in_sizes, int n_in,
                              void* d_out, int out_size, void* d_ws, size_t ws_size,
                              hipStream_t stream) {
    (void)in_sizes; (void)n_in; (void)out_size; (void)ws_size;
    const float* x      = (const float*)d_in[0];
    const float* qkv_w  = (const float*)d_in[1];
    const float* qkv_b  = (const float*)d_in[2];
    const float* proj_w = (const float*)d_in[3];
    const float* proj_b = (const float*)d_in[4];

    _Float16* ws = (_Float16*)d_ws;
    size_t o = 0;
    _Float16* xh     = ws + o; o += (size_t)M_ROWS * D_MODEL;
    _Float16* wqt    = ws + o; o += (size_t)D_MODEL * N_QKV;    // [3C, C] = W^T
    _Float16* wpt    = ws + o; o += (size_t)D_MODEL * D_MODEL;  // [C, C] = W^T
    _Float16* qkvh   = ws + o; o += (size_t)M_ROWS * N_QKV;
    _Float16* qheads = ws + o; o += (size_t)M_ROWS * D_MODEL;
    _Float16* kheads = ws + o; o += (size_t)M_ROWS * D_MODEL;
    _Float16* vtr    = ws + o; o += (size_t)M_ROWS * D_MODEL;   // [B,H,Dh,T]
    _Float16* attnh  = ws + o; o += (size_t)M_ROWS * D_MODEL;

    const int nx8 = M_ROWS * D_MODEL / 8;
    cast_f32_to_f16_v8<<<(nx8 + 255) / 256, 256, 0, stream>>>(x, xh, nx8);
    cast_transpose_f16<<<dim3(D_MODEL / 64, N_QKV / 64), 256, 0, stream>>>(
        qkv_w, wqt, D_MODEL, N_QKV);
    cast_transpose_f16<<<dim3(D_MODEL / 64, D_MODEL / 64), 256, 0, stream>>>(
        proj_w, wpt, D_MODEL, D_MODEL);

    gemm_wmma_f16<_Float16><<<dim3(M_ROWS / BM, N_QKV / BN), 256, 0, stream>>>(
        xh, wqt, qkv_b, qkvh, M_ROWS, N_QKV, D_MODEL);

    const int nqk8 = M_ROWS * 2 * D_MODEL / 8;
    split_qk<<<(nqk8 + 255) / 256, 256, 0, stream>>>(qkvh, qheads, kheads);
    transpose_v<<<dim3(BATCH * N_HEADS, SEQ_T / 64), 256, 0, stream>>>(qkvh, vtr);

    attn_wmma<<<dim3(BATCH * N_HEADS, SEQ_T / 128), 128, 0, stream>>>(
        qheads, kheads, vtr, attnh);

    gemm_wmma_f16<float><<<dim3(M_ROWS / BM, D_MODEL / BN), 256, 0, stream>>>(
        attnh, wpt, proj_b, (float*)d_out, M_ROWS, D_MODEL, D_MODEL);
}
